// Captioner_31439160607449
// MI455X (gfx1250) — matrix operand order repaired and checked by reading, not compiled, not run on hardware
//
#include <hip/hip_runtime.h>

// ---------------------------------------------------------------------------
// Autoregressive GRU caption decoder for MI455X (gfx1250, wave32, WMMA).
//   V=100000, H=1280, E=300, L=32.
// Dominant cost: W_cls GEMV (512 MB f32 / step, 32 steps). We convert W_cls
// to bf16 once per launch (into d_ws) in a WMMA-friendly swizzled tiling and
// run the classifier GEMV with v_wmma_f32_16x16x32_bf16, halving HBM traffic
// (16.4 GB -> 8.2 GB => ~0.35 ms at 23.3 TB/s).
// ---------------------------------------------------------------------------

#define V_SZ   100000
#define H_SZ   1280
#define E_SZ   300
#define L_SZ   32
#define H3_SZ  (3 * H_SZ)            // 3840
#define NTILES (V_SZ / 16)           // 6250 16-row WMMA tiles
#define NKC    (H_SZ / 32)           // 40 K-chunks of 32

// workspace byte offsets (all 64B-aligned)
#define WS_FLAGS 0u
#define WS_H     1024u
#define WS_EMB   6144u
#define WS_GI    7424u
#define WS_GH    22784u
#define WS_HBF   38144u
#define WS_CAND  40960u
#define WS_WP    262144u
#define WP_BYTES (256000000ull)      // V*H*2

typedef __bf16 bf16_t;
typedef bf16_t v16bf __attribute__((ext_vector_type(16)));
typedef float  v8f   __attribute__((ext_vector_type(8)));

struct Cand { float v; int i; };

__device__ __forceinline__ unsigned short f32_to_bf16_rne(float f) {
  unsigned u = __float_as_uint(f);
  unsigned r = u + 0x7FFFu + ((u >> 16) & 1u);   // round-to-nearest-even
  return (unsigned short)(r >> 16);
}
__device__ __forceinline__ unsigned pk_bf16(float a, float b) {
  return (unsigned)f32_to_bf16_rne(a) | ((unsigned)f32_to_bf16_rne(b) << 16);
}
__device__ __forceinline__ float sigmoidf_(float x) { return 1.0f / (1.0f + expf(-x)); }

// ---------------------------------------------------------------------------
// Init per-call state: h = hidden0, emb = start_emb, done = 0.
// ---------------------------------------------------------------------------
__global__ void init_state(const float* __restrict__ hidden0,
                           const float* __restrict__ start_emb,
                           float* __restrict__ h, float* __restrict__ emb,
                           int* __restrict__ flags) {
  int tid = threadIdx.x;
  for (int j = tid; j < H_SZ; j += 1024) h[j] = hidden0[j];
  for (int j = tid; j < E_SZ; j += 1024) emb[j] = start_emb[j];
  if (tid == 0) flags[0] = 0;
}

// ---------------------------------------------------------------------------
// Convert W_cls (f32 row-major [V,H]) -> bf16, swizzled into 16x32 A tiles
// in contiguous k order (k = 16*(lane>>4) + s) so each GEMV lane does ONE contiguous
// 32-byte load per WMMA. Per (tile g, kchunk kc, lane):
//   bytes[ 0..15] = W[g*16 + (lane&15)][kc*32 + (lane>>4)*16     .. +7]  (VGPR0-3)
//   bytes[16..31] = W[g*16 + (lane&15)][kc*32 + (lane>>4)*16 + 8 .. +7]  (VGPR4-7)
// One thread emits one 16-byte half-chunk. total = NTILES*NKC*32*2 = 16e6.
// ---------------------------------------------------------------------------
__global__ void convert_wcls(const float* __restrict__ W,
                             unsigned short* __restrict__ Wp) {
  unsigned cid = blockIdx.x * 256u + threadIdx.x;      // [0, 16,000,000)
  unsigned u    = cid >> 1;
  int      h2   = (int)(cid & 1u);
  int      lane = (int)(u & 31u);
  unsigned t40  = u >> 5;                               // tile*NKC + kc
  int      kc   = (int)(t40 % NKC);
  int      g    = (int)(t40 / NKC);
  int      row  = g * 16 + (lane & 15);
  int      k0   = kc * 32 + ((lane >> 4) << 4) + (h2 << 3);
  const float4* src = (const float4*)(W + (size_t)row * H_SZ + k0);
  float4 f0 = src[0];
  float4 f1 = src[1];
  uint4 out;
  out.x = pk_bf16(f0.x, f0.y);
  out.y = pk_bf16(f0.z, f0.w);
  out.z = pk_bf16(f1.x, f1.y);
  out.w = pk_bf16(f1.z, f1.w);
  *(uint4*)(Wp + (size_t)cid * 8) = out;
}

// ---------------------------------------------------------------------------
// GRU gate GEMVs: gi[i] = W_ih[i,:]@emb + b_ih[i]; gh[i] = W_hh[i,:]@h + b_hh[i]
// One wave per output row; 480 blocks x 256 threads = 3840 waves.
// ---------------------------------------------------------------------------
__global__ void gru_gates(const float* __restrict__ W_ih,
                          const float* __restrict__ W_hh,
                          const float* __restrict__ b_ih,
                          const float* __restrict__ b_hh,
                          const float* __restrict__ h,
                          const float* __restrict__ emb,
                          float* __restrict__ gi, float* __restrict__ gh) {
  int wid  = threadIdx.x >> 5;
  int lane = threadIdx.x & 31;
  int row  = blockIdx.x * 8 + wid;                 // < 3840 by construction

  const float* wi = W_ih + (size_t)row * E_SZ;
  float si = 0.0f;
  for (int k = lane; k < E_SZ; k += 32) si += wi[k] * emb[k];

  const float4* wh = (const float4*)(W_hh + (size_t)row * H_SZ);
  const float4* hv = (const float4*)h;
  float sh = 0.0f;
#pragma unroll
  for (int j = 0; j < 10; ++j) {                   // 10*32*4 = 1280
    float4 a = wh[lane + j * 32];
    float4 b = hv[lane + j * 32];
    sh += a.x * b.x + a.y * b.y + a.z * b.z + a.w * b.w;
  }
  for (int m = 16; m >= 1; m >>= 1) {
    si += __shfl_xor(si, m);
    sh += __shfl_xor(sh, m);
  }
  if (lane == 0) {
    gi[row] = si + b_ih[row];
    gh[row] = sh + b_hh[row];
  }
}

// ---------------------------------------------------------------------------
// GRU combine: r/z/n gates, done-masked h update; writes h (f32), hiddens[t],
// and bf16 h for the WMMA classifier. 1280 threads.
// ---------------------------------------------------------------------------
__global__ void gru_combine(const float* __restrict__ gi,
                            const float* __restrict__ gh,
                            float* __restrict__ h,
                            unsigned short* __restrict__ hbf,
                            const int* __restrict__ flags,
                            float* __restrict__ hid_out) {
  int j = blockIdx.x * blockDim.x + threadIdx.x;
  if (j >= H_SZ) return;
  int   done = flags[0];
  float hj = h[j];
  float r  = sigmoidf_(gi[j] + gh[j]);
  float z  = sigmoidf_(gi[H_SZ + j] + gh[H_SZ + j]);
  float n  = tanhf(gi[2 * H_SZ + j] + r * gh[2 * H_SZ + j]);
  float hn = (1.0f - z) * n + z * hj;
  float hnew = done ? hj : hn;
  h[j]       = hnew;
  hid_out[j] = hnew;
  hbf[j]     = f32_to_bf16_rne(hnew);
}

// ---------------------------------------------------------------------------
// Classifier GEMV via v_wmma_f32_16x16x32_bf16.
// Wave g handles vocab rows [16g, 16g+16). A = swizzled W_cls tile (one
// coalesced 32B load/lane/WMMA). B = h broadcast across the 16 columns:
// matching A's contiguous k order, lanes 0-15 carry K=0..15, lanes 16-31 carry K=16..31 of the
// chunk -> a single aligned contiguous v16bf broadcast load from hbf.
// D columns are identical => c[0..7] hold the 8 row dot-products per half.
// Emit one (max,argmax) candidate per wave (first-index tie-break).
// ---------------------------------------------------------------------------
__global__ void cls_wmma(const unsigned short* __restrict__ Wp,
                         const unsigned short* __restrict__ hbf,
                         const float* __restrict__ b_cls,
                         Cand* __restrict__ cand, int ntiles) {
  int wid  = threadIdx.x >> 5;
  int lane = threadIdx.x & 31;
  int g    = blockIdx.x * 8 + wid;
  if (g >= ntiles) return;                     // wave-uniform; EXEC stays all-1s
  int laneHi = lane >> 4;

  const unsigned short* tile = Wp + (size_t)g * (NKC * 512);  // 512 u16 per kc
  v8f c = {};
#pragma unroll 4
  for (int kc = 0; kc < NKC; ++kc) {
    v16bf a = *(const v16bf*)(tile + kc * 512 + lane * 16);
    v16bf b = *(const v16bf*)(hbf + kc * 32 + laneHi * 16);
    if (kc + 1 < NKC)
      __builtin_prefetch(tile + (kc + 1) * 512 + lane * 16, 0, 1);
    c = __builtin_amdgcn_wmma_f32_16x16x32_bf16(
            /*neg_a=*/false, a, /*neg_b=*/false, b,
            /*c_mod=*/(short)0, c, /*reuse_a=*/false, /*reuse_b=*/false);
  }

  // rows for this half-wave: lanes 0-15 -> M=0..7, lanes 16-31 -> M=8..15
  int   rbase = g * 16 + laneHi * 8;
  float bv = -3.4e38f;
  int   bi = rbase;
#pragma unroll
  for (int j = 0; j < 8; ++j) {
    float v = c[j] + b_cls[rbase + j];
    if (v > bv) { bv = v; bi = rbase + j; }     // strict > keeps first index
  }
  float ov = __shfl(bv, 16);
  int   oi = __shfl(bi, 16);
  if (lane == 0) {
    if (ov > bv || (ov == bv && oi < bi)) { bv = ov; bi = oi; }
    cand[g].v = bv;
    cand[g].i = bi;
  }
}

// ---------------------------------------------------------------------------
// Fallback f32 classifier (used only if d_ws cannot hold the bf16 copy).
// One wave per vocab row; per-block argmax over its 8 rows.
// ---------------------------------------------------------------------------
__global__ void cls_f32(const float* __restrict__ W,
                        const float* __restrict__ h,
                        const float* __restrict__ b_cls,
                        Cand* __restrict__ cand) {
  __shared__ float sv[8];
  __shared__ int   si[8];
  int wid  = threadIdx.x >> 5;
  int lane = threadIdx.x & 31;
  int row  = blockIdx.x * 8 + wid;
  float s = 0.0f;
  if (row < V_SZ) {
    const float4* wr = (const float4*)(W + (size_t)row * H_SZ);
    const float4* hv = (const float4*)h;
#pragma unroll
    for (int j = 0; j < 10; ++j) {
      float4 a = wr[lane + j * 32];
      float4 b = hv[lane + j * 32];
      s += a.x * b.x + a.y * b.y + a.z * b.z + a.w * b.w;
    }
  }
  for (int m = 16; m >= 1; m >>= 1) s += __shfl_xor(s, m);
  if (lane == 0) {
    if (row < V_SZ) { sv[wid] = s + b_cls[row]; si[wid] = row; }
    else            { sv[wid] = -3.4e38f;        si[wid] = 0x7fffffff; }
  }
  __syncthreads();
  if (threadIdx.x == 0) {
    float bv = sv[0]; int bi = si[0];
    for (int w = 1; w < 8; ++w)
      if (sv[w] > bv || (sv[w] == bv && si[w] < bi)) { bv = sv[w]; bi = si[w]; }
    cand[blockIdx.x].v = bv;
    cand[blockIdx.x].i = bi;
  }
}

// ---------------------------------------------------------------------------
// Final argmax over candidates + token/done/emb update. One 1024-thread block.
// ---------------------------------------------------------------------------
__global__ void finalize_step(const Cand* __restrict__ cand, int ncand,
                              int* __restrict__ flags, float* __restrict__ emb,
                              const float* __restrict__ vocab,
                              int* __restrict__ tok_out, int t) {
  __shared__ float sv[32];
  __shared__ int   si[32];
  int tid = threadIdx.x, lane = tid & 31, wid = tid >> 5;
  int done_old = flags[0];

  float bv = -3.4e38f;
  int   bi = 0x7fffffff;
  for (int i = tid; i < ncand; i += 1024) {
    float v = cand[i].v;
    int   ix = cand[i].i;
    if (v > bv || (v == bv && ix < bi)) { bv = v; bi = ix; }
  }
  for (int m = 16; m >= 1; m >>= 1) {
    float ov = __shfl_xor(bv, m);
    int   oi = __shfl_xor(bi, m);
    if (ov > bv || (ov == bv && oi < bi)) { bv = ov; bi = oi; }
  }
  if (lane == 0) { sv[wid] = bv; si[wid] = bi; }
  __syncthreads();
  if (wid == 0) {
    bv = sv[lane]; bi = si[lane];
    for (int m = 16; m >= 1; m >>= 1) {
      float ov = __shfl_xor(bv, m);
      int   oi = __shfl_xor(bi, m);
      if (ov > bv || (ov == bv && oi < bi)) { bv = ov; bi = oi; }
    }
    if (lane == 0) { sv[0] = bv; si[0] = bi; }
  }
  __syncthreads();

  int pred     = si[0];
  int token    = done_old ? 0 : pred;
  int done_new = (done_old || (pred == 0)) ? 1 : 0;
  if (tid == 0) { tok_out[t] = token; flags[0] = done_new; }
  if (!done_new)
    for (int j = tid; j < E_SZ; j += 1024)
      emb[j] = vocab[(size_t)pred * E_SZ + j];
}

// ---------------------------------------------------------------------------
extern "C" void kernel_launch(void* const* d_in, const int* in_sizes, int n_in,
                              void* d_out, int out_size, void* d_ws, size_t ws_size,
                              hipStream_t stream) {
  (void)in_sizes; (void)n_in; (void)out_size;
  const float* hidden0   = (const float*)d_in[0];
  const float* start_emb = (const float*)d_in[1];
  const float* W_ih      = (const float*)d_in[2];
  const float* W_hh      = (const float*)d_in[3];
  const float* b_ih      = (const float*)d_in[4];
  const float* b_hh      = (const float*)d_in[5];
  const float* W_cls     = (const float*)d_in[6];
  const float* b_cls     = (const float*)d_in[7];
  const float* vocab     = (const float*)d_in[8];

  char* ws = (char*)d_ws;
  int*            flags = (int*)(ws + WS_FLAGS);
  float*          h     = (float*)(ws + WS_H);
  float*          emb   = (float*)(ws + WS_EMB);
  float*          gi    = (float*)(ws + WS_GI);
  float*          gh    = (float*)(ws + WS_GH);
  unsigned short* hbf   = (unsigned short*)(ws + WS_HBF);
  Cand*           cand  = (Cand*)(ws + WS_CAND);
  unsigned short* Wp    = (unsigned short*)(ws + WS_WP);

  int*   tok_out = (int*)d_out;                 // tokens [L] (int32)
  float* hid_out = (float*)d_out + L_SZ;        // hiddens [L,H] (f32)

  const bool use_bf16 = ws_size >= (size_t)WS_WP + WP_BYTES;

  init_state<<<1, 1024, 0, stream>>>(hidden0, start_emb, h, emb, flags);
  if (use_bf16)
    convert_wcls<<<62500, 256, 0, stream>>>(W_cls, Wp);   // 16e6 threads

  for (int t = 0; t < L_SZ; ++t) {
    gru_gates<<<H3_SZ / 8, 256, 0, stream>>>(W_ih, W_hh, b_ih, b_hh,
                                             h, emb, gi, gh);
    gru_combine<<<(H_SZ + 255) / 256, 256, 0, stream>>>(gi, gh, h, hbf, flags,
                                                        hid_out + (size_t)t * H_SZ);
    int ncand;
    if (use_bf16) {
      cls_wmma<<<(NTILES + 7) / 8, 256, 0, stream>>>(Wp, hbf, b_cls, cand, NTILES);
      ncand = NTILES;                                     // 6250
    } else {
      cls_f32<<<V_SZ / 8, 256, 0, stream>>>(W_cls, h, b_cls, cand);
      ncand = V_SZ / 8;                                   // 12500
    }
    finalize_step<<<1, 1024, 0, stream>>>(cand, ncand, flags, emb, vocab,
                                          tok_out, t);
  }
}
